// GPT2Block_38654705664175
// MI455X (gfx1250) — matrix operand order repaired and checked by reading, not compiled, not run on hardware
//
#include <hip/hip_runtime.h>
#include <hip/hip_bf16.h>
#include <math.h>

// ---------------------------------------------------------------------------
// GPT-2 block forward for MI455X (gfx1250, wave32, WMMA).
// B=2, T=2048, C=1024, H=16, Dh=64, DFF=4096.  All GEMMs in bf16 WMMA w/ f32 acc.
// ---------------------------------------------------------------------------

typedef __attribute__((ext_vector_type(16))) __bf16 v16bf;
typedef __attribute__((ext_vector_type(8)))  __bf16 v8bf;
typedef __attribute__((ext_vector_type(8)))  float  v8f;

union ABFrag { v16bf v; v8bf h[2]; };

static __device__ inline __bf16 f2bf(float f) {
    unsigned u = __builtin_bit_cast(unsigned, f);
    u += 0x7fffu + ((u >> 16) & 1u);           // round to nearest even
    unsigned short hs = (unsigned short)(u >> 16);
    return __builtin_bit_cast(__bf16, hs);
}

// ---------------------------------------------------------------------------
// Weight convert + transpose: w[K,N] f32 -> wT[N,K] bf16
// ---------------------------------------------------------------------------
__global__ void conv_tr_kernel(const float* __restrict__ w, __bf16* __restrict__ wT,
                               long K, long N) {
    long idx = (long)blockIdx.x * blockDim.x + threadIdx.x;
    if (idx >= K * N) return;
    long n = idx / K;
    long k = idx % K;
    wT[idx] = f2bf(w[k * N + n]);
}

// ---------------------------------------------------------------------------
// LayerNorm: f32 [rows, 1024] -> bf16 [rows, 1024]. One block per row.
// ---------------------------------------------------------------------------
__global__ __launch_bounds__(256) void layernorm_kernel(
    const float* __restrict__ x, const float* __restrict__ g,
    const float* __restrict__ b, __bf16* __restrict__ out) {
    const int C = 1024;
    const int row = blockIdx.x;
    const int tid = threadIdx.x;
    __shared__ float s1[256], s2[256];

    float4 v = *(const float4*)(x + (size_t)row * C + tid * 4);
    float sum = v.x + v.y + v.z + v.w;
    float sq  = v.x * v.x + v.y * v.y + v.z * v.z + v.w * v.w;
    s1[tid] = sum; s2[tid] = sq;
    __syncthreads();
    for (int s = 128; s > 0; s >>= 1) {
        if (tid < s) { s1[tid] += s1[tid + s]; s2[tid] += s2[tid + s]; }
        __syncthreads();
    }
    float mu  = s1[0] * (1.0f / C);
    float var = s2[0] * (1.0f / C) - mu * mu;
    float rs  = rsqrtf(var + 1e-5f);

    float vals[4] = {v.x, v.y, v.z, v.w};
    #pragma unroll
    for (int k = 0; k < 4; k++) {
        int c = tid * 4 + k;
        out[(size_t)row * C + c] = f2bf((vals[k] - mu) * rs * g[c] + b[c]);
    }
}

// ---------------------------------------------------------------------------
// WMMA GEMM: C[M,N] = A[M,K](bf16) * Bt[N,K](bf16)^T + bias, with epilogue.
// Each wave computes a 64x64 tile: 4x4 grid of 16x16 WMMA accumulators.
// epi: 0 = bias -> bf16 out
//      1 = bias + resid(f32) -> f32 out
//      2 = bias + exact GELU -> bf16 out
// ---------------------------------------------------------------------------
__global__ __launch_bounds__(128) void gemm_wmma_bf16_kernel(
    const __bf16* __restrict__ A, const __bf16* __restrict__ Bt,
    const float* __restrict__ bias, const float* __restrict__ resid,
    void* __restrict__ out, int M, int N, int K, int epi) {
    const int lane = threadIdx.x & 31;
    const int l16  = lane & 15;
    const int half = lane >> 4;
    const int tilesN = N >> 6;
    const int w = blockIdx.x * 4 + (threadIdx.x >> 5);
    if (w >= (M >> 6) * tilesN) return;
    const int tm = w / tilesN;
    const int tn = w % tilesN;

    v8f acc[4][4];
    #pragma unroll
    for (int i = 0; i < 4; i++)
        #pragma unroll
        for (int j = 0; j < 4; j++)
            #pragma unroll
            for (int r = 0; r < 8; r++) acc[i][j][r] = 0.0f;

    const __bf16* arow[4];
    const __bf16* brow[4];
    #pragma unroll
    for (int i = 0; i < 4; i++)
        arow[i] = A + (size_t)(tm * 64 + i * 16 + l16) * K;
    #pragma unroll
    for (int j = 0; j < 4; j++)
        brow[j] = Bt + (size_t)(tn * 64 + j * 16 + l16) * K;

    for (int k0 = 0; k0 < K; k0 += 32) {
        ABFrag a[4], b[4];
        #pragma unroll
        for (int i = 0; i < 4; i++) {
            // A 16x32 bf16 fragment: lanes 0-15 row-major, K packed 8+8 per half
            a[i].h[0] = *(const v8bf*)(arow[i] + k0 + half * 8);
            a[i].h[1] = *(const v8bf*)(arow[i] + k0 + 16 + half * 8);
        }
        #pragma unroll
        for (int j = 0; j < 4; j++) {
            // B 32x16 bf16 fragment from transposed weights: K packed 8+8 per half
            b[j].h[0] = *(const v8bf*)(brow[j] + k0 + half * 8);
            b[j].h[1] = *(const v8bf*)(brow[j] + k0 + 16 + half * 8);
        }
        #pragma unroll
        for (int i = 0; i < 4; i++)
            #pragma unroll
            for (int j = 0; j < 4; j++)
                acc[i][j] = __builtin_amdgcn_wmma_f32_16x16x32_bf16(
                    false, a[i].v, false, b[j].v, (short)0, acc[i][j], false, false);
    }

    #pragma unroll
    for (int j = 0; j < 4; j++) {
        const int col = tn * 64 + j * 16 + l16;
        const float bb = bias[col];
        #pragma unroll
        for (int i = 0; i < 4; i++) {
            #pragma unroll
            for (int r = 0; r < 8; r++) {
                const int row = tm * 64 + i * 16 + r + 8 * half;
                const size_t idx = (size_t)row * N + col;
                float v = acc[i][j][r] + bb;
                if (epi == 0) {
                    ((__bf16*)out)[idx] = f2bf(v);
                } else if (epi == 1) {
                    ((float*)out)[idx] = v + resid[idx];
                } else {
                    v = 0.5f * v * (1.0f + erff(v * 0.70710678118654752f));
                    ((__bf16*)out)[idx] = f2bf(v);
                }
            }
        }
    }
}

// ---------------------------------------------------------------------------
// Flash attention: one wave per (b, h, 16-query tile). Causal, Dh=64.
// qkv bf16 [B*T, 3072]: q @ h*64, k @ 1024+h*64, v @ 2048+h*64.
// S via WMMA (K rows contiguous in d -> b128 fragments), online softmax,
// P transposed C-layout->A-layout through LDS, O += P*V via WMMA (32-key step).
// ---------------------------------------------------------------------------
__global__ __launch_bounds__(32) void attn_kernel(
    const __bf16* __restrict__ qkv, __bf16* __restrict__ att) {
    const int T = 2048, C3 = 3072, CO = 1024;
    const int mt = blockIdx.x;     // query tile (16 rows)
    const int h  = blockIdx.y;
    const int bb = blockIdx.z;
    const int lane = threadIdx.x & 31;
    const int l16  = lane & 15;
    const int half = lane >> 4;

    __shared__ __bf16 pls[16 * 32];

    // Q fragments (16x64), loaded once: two 16x32 A-fragments
    ABFrag qf[2];
    {
        const __bf16* qp = qkv + (size_t)(bb * T + mt * 16 + l16) * C3 + h * 64;
        #pragma unroll
        for (int d = 0; d < 2; d++) {
            qf[d].h[0] = *(const v8bf*)(qp + d * 32 + half * 8);
            qf[d].h[1] = *(const v8bf*)(qp + d * 32 + 16 + half * 8);
        }
    }

    v8f o[4];
    #pragma unroll
    for (int c = 0; c < 4; c++)
        #pragma unroll
        for (int r = 0; r < 8; r++) o[c][r] = 0.0f;
    float mrow[8], lrow[8];
    #pragma unroll
    for (int r = 0; r < 8; r++) { mrow[r] = -1e30f; lrow[r] = 0.0f; }

    for (int j0 = 0; j0 < (mt + 1) * 16; j0 += 32) {
        // ---- S = Q * K^T for 32 keys (two 16-key subtiles) ----
        v8f s[2];
        #pragma unroll
        for (int st = 0; st < 2; st++) {
            #pragma unroll
            for (int r = 0; r < 8; r++) s[st][r] = 0.0f;
            const __bf16* kp = qkv + (size_t)(bb * T + j0 + st * 16 + l16) * C3
                               + 1024 + h * 64;
            #pragma unroll
            for (int d = 0; d < 2; d++) {
                ABFrag kb;
                kb.h[0] = *(const v8bf*)(kp + d * 32 + half * 8);
                kb.h[1] = *(const v8bf*)(kp + d * 32 + 16 + half * 8);
                s[st] = __builtin_amdgcn_wmma_f32_16x16x32_bf16(
                    false, qf[d].v, false, kb.v, (short)0, s[st], false, false);
            }
        }
        // ---- scale + causal mask ----
        const int qg = mt * 16 + 8 * half;   // + r below
        #pragma unroll
        for (int st = 0; st < 2; st++) {
            const int kg = j0 + st * 16 + l16;
            #pragma unroll
            for (int r = 0; r < 8; r++) {
                float v = s[st][r] * 0.125f;  // 1/sqrt(64)
                if (kg > qg + r) v = -1e30f;
                s[st][r] = v;
            }
        }
        // ---- online softmax (row stats across 16 lanes of each half) ----
        float alpha[8];
        #pragma unroll
        for (int r = 0; r < 8; r++) {
            float mx = fmaxf(s[0][r], s[1][r]);
            mx = fmaxf(mx, __shfl_xor(mx, 1, 32));
            mx = fmaxf(mx, __shfl_xor(mx, 2, 32));
            mx = fmaxf(mx, __shfl_xor(mx, 4, 32));
            mx = fmaxf(mx, __shfl_xor(mx, 8, 32));
            float mn = fmaxf(mrow[r], mx);
            float al = __expf(mrow[r] - mn);
            float p0 = __expf(s[0][r] - mn);
            float p1 = __expf(s[1][r] - mn);
            s[0][r] = p0; s[1][r] = p1;
            float ps = p0 + p1;
            ps += __shfl_xor(ps, 1, 32);
            ps += __shfl_xor(ps, 2, 32);
            ps += __shfl_xor(ps, 4, 32);
            ps += __shfl_xor(ps, 8, 32);
            lrow[r] = lrow[r] * al + ps;
            mrow[r] = mn;
            alpha[r] = al;
        }
        #pragma unroll
        for (int c = 0; c < 4; c++)
            #pragma unroll
            for (int r = 0; r < 8; r++) o[c][r] *= alpha[r];

        // ---- P: C-layout -> A-layout via LDS (16 rows x 32 keys, bf16) ----
        #pragma unroll
        for (int st = 0; st < 2; st++)
            #pragma unroll
            for (int r = 0; r < 8; r++)
                pls[(r + 8 * half) * 32 + st * 16 + l16] = f2bf(s[st][r]);
        asm volatile("s_wait_dscnt 0x0" ::: "memory");
        ABFrag pf;
        pf.h[0] = *(const v8bf*)&pls[l16 * 32 + half * 8];
        pf.h[1] = *(const v8bf*)&pls[l16 * 32 + 16 + half * 8];

        // ---- O += P * V  (V fragments gathered: coalesced across lanes) ----
        #pragma unroll
        for (int c = 0; c < 4; c++) {
            ABFrag vb;
            const __bf16* vp = qkv + 2048 + h * 64 + c * 16 + l16;
            #pragma unroll
            for (int e = 0; e < 16; e++) {
                int kk = j0 + (e < 8 ? half * 8 + e : 16 + half * 8 + (e - 8));
                vb.v[e] = vp[(size_t)(bb * T + kk) * C3];
            }
            o[c] = __builtin_amdgcn_wmma_f32_16x16x32_bf16(
                false, pf.v, false, vb.v, (short)0, o[c], false, false);
        }
    }

    // ---- normalize + store bf16 ----
    #pragma unroll
    for (int c = 0; c < 4; c++) {
        #pragma unroll
        for (int r = 0; r < 8; r++) {
            const int row = bb * T + mt * 16 + r + 8 * half;
            const int col = h * 64 + c * 16 + l16;
            att[(size_t)row * CO + col] = f2bf(o[c][r] / lrow[r]);
        }
    }
}

// ---------------------------------------------------------------------------
// Host launcher
// ---------------------------------------------------------------------------
extern "C" void kernel_launch(void* const* d_in, const int* in_sizes, int n_in,
                              void* d_out, int out_size, void* d_ws, size_t ws_size,
                              hipStream_t stream) {
    (void)in_sizes; (void)n_in; (void)out_size; (void)ws_size;
    const float* x       = (const float*)d_in[0];
    const float* ln1_g   = (const float*)d_in[1];
    const float* ln1_b   = (const float*)d_in[2];
    const float* w_attn  = (const float*)d_in[3];
    const float* b_attn  = (const float*)d_in[4];
    const float* w_aproj = (const float*)d_in[5];
    const float* b_aproj = (const float*)d_in[6];
    const float* ln2_g   = (const float*)d_in[7];
    const float* ln2_b   = (const float*)d_in[8];
    const float* w_fc    = (const float*)d_in[9];
    const float* b_fc    = (const float*)d_in[10];
    const float* w_mproj = (const float*)d_in[11];
    const float* b_mproj = (const float*)d_in[12];

    const long M = 4096;   // B*T
    char* ws = (char*)d_ws;
    size_t off = 0;
    auto alloc = [&](size_t bytes) { char* p = ws + off; off += (bytes + 255) & ~size_t(255); return p; };
    __bf16* wT_attn  = (__bf16*)alloc(3072L * 1024 * 2);
    __bf16* wT_aproj = (__bf16*)alloc(1024L * 1024 * 2);
    __bf16* wT_fc    = (__bf16*)alloc(4096L * 1024 * 2);
    __bf16* wT_mproj = (__bf16*)alloc(1024L * 4096 * 2);
    __bf16* h_bf     = (__bf16*)alloc(M * 1024 * 2);
    __bf16* qkv      = (__bf16*)alloc(M * 3072 * 2);
    __bf16* attb     = (__bf16*)alloc(M * 1024 * 2);
    float*  x2       = (float*) alloc(M * 1024 * 4);
    __bf16* ff       = (__bf16*)alloc(M * 4096 * 2);

    auto cdiv = [](long a, long b) { return (unsigned)((a + b - 1) / b); };

    // weights -> bf16 transposed [N,K]
    conv_tr_kernel<<<cdiv(1024L * 3072, 256), 256, 0, stream>>>(w_attn,  wT_attn,  1024, 3072);
    conv_tr_kernel<<<cdiv(1024L * 1024, 256), 256, 0, stream>>>(w_aproj, wT_aproj, 1024, 1024);
    conv_tr_kernel<<<cdiv(1024L * 4096, 256), 256, 0, stream>>>(w_fc,    wT_fc,    1024, 4096);
    conv_tr_kernel<<<cdiv(4096L * 1024, 256), 256, 0, stream>>>(w_mproj, wT_mproj, 4096, 1024);

    // h1 = LN1(x)
    layernorm_kernel<<<(unsigned)M, 256, 0, stream>>>(x, ln1_g, ln1_b, h_bf);

    // qkv = h1 @ w_attn + b_attn        [4096,3072] bf16
    gemm_wmma_bf16_kernel<<<(64 * 48) / 4, 128, 0, stream>>>(
        h_bf, wT_attn, b_attn, nullptr, qkv, 4096, 3072, 1024, 0);

    // attention -> attb [4096,1024] bf16
    attn_kernel<<<dim3(128, 16, 2), 32, 0, stream>>>(qkv, attb);

    // x2 = x + attb @ w_aproj + b_aproj  (f32)
    gemm_wmma_bf16_kernel<<<(64 * 16) / 4, 128, 0, stream>>>(
        attb, wT_aproj, b_aproj, x, x2, 4096, 1024, 1024, 1);

    // h2 = LN2(x2)
    layernorm_kernel<<<(unsigned)M, 256, 0, stream>>>(x2, ln2_g, ln2_b, h_bf);

    // ff = gelu(h2 @ w_fc + b_fc)        [4096,4096] bf16
    gemm_wmma_bf16_kernel<<<(64 * 64) / 4, 128, 0, stream>>>(
        h_bf, wT_fc, b_fc, nullptr, ff, 4096, 4096, 1024, 2);

    // out = x2 + ff @ w_mproj + b_mproj  (f32)
    gemm_wmma_bf16_kernel<<<(64 * 16) / 4, 128, 0, stream>>>(
        ff, wT_mproj, b_mproj, x2, d_out, 4096, 1024, 4096, 1);
}
